// CINNv2_22419729285718
// MI455X (gfx1250) — matrix operand order repaired and checked by reading, not compiled, not run on hardware
//
#include <hip/hip_runtime.h>
#include <hip/hip_bf16.h>
#include <math.h>

// ---------- types ----------
typedef __bf16 bf16_t;
typedef __attribute__((ext_vector_type(16))) __bf16        v16bf;
typedef __attribute__((ext_vector_type(4)))  __bf16        v4bf;
typedef __attribute__((ext_vector_type(8)))  float         v8f;
typedef __attribute__((ext_vector_type(4)))  float         v4f;
typedef __attribute__((ext_vector_type(4)))  unsigned int  v4u;

// ---------- problem constants ----------
#define D_FEAT  64
#define HID     512
#define DC      128
#define KBLK    12
#define L1      32
#define NIN     160        // L + DC = 32 + 128
#define BT      128        // batch rows per workgroup
#define NWAVE   8
#define NTHR    256

// ---------- LDS layout (dynamic segment), all fragment chunks 16B aligned ----------
#define W1_STRIDE  336     // bytes per W1 row (160 bf16 + pad, 16B multiple)
#define W2_STRIDE  1056    // bytes per W2 row (512 bf16 + pad, 16B multiple)
#define ACT_STRIDE 336     // bytes per activation row (160 bf16 + pad)
#define OFF_W1   0
#define SZ_W1    (HID * W1_STRIDE)          // 172032... (see below: recomputed)
#define OFF_W2   (OFF_W1 + SZ_W1)
#define SZ_W2    (64 * W2_STRIDE)           // 67584
#define OFF_ACT  (OFF_W2 + SZ_W2)
#define SZ_ACT   (BT * ACT_STRIDE)          // 43008
#define OFF_X    (OFF_ACT + SZ_ACT)
#define SZ_X     (BT * D_FEAT * 4)          // 32768
#define OFF_HS   (OFF_X + SZ_X)
#define SZ_HS    (NWAVE * 1024)             // per-wave 16x32 bf16 hidden scratch
#define SMEM_DYN (OFF_HS + SZ_HS)

union Frag { v16bf v; v4u q[2]; };

__device__ __forceinline__ v8f wmma_bf16(v16bf a, v16bf b, v8f c) {
  // D = A(16x32 bf16) x B(32x16 bf16) + C(16x16 f32)
  return __builtin_amdgcn_wmma_f32_16x16x32_bf16(
      /*neg_a=*/false, a, /*neg_b=*/false, b,
      /*c_mod=*/(short)0, c, /*reuse_a=*/false, /*reuse_b=*/false);
}

// Cooperatively stage one MLP's weights (f32 global -> bf16 LDS) + biases.
__device__ __forceinline__ void stage_weights(char* sm,
                                              const float* W1g, const float* b1g,
                                              const float* W2g, const float* b2g,
                                              float* s_b1, float* s_b2, int tid) {
  // W1: [512][160] row-major
  const v4f* src1 = (const v4f*)W1g;
  for (int i = tid; i < (HID * NIN) / 4; i += NTHR) {
    int e = i * 4;
    int n = e / NIN, c = e % NIN;
    v4f f = src1[i];
    v4bf h = { (bf16_t)f[0], (bf16_t)f[1], (bf16_t)f[2], (bf16_t)f[3] };
    *(v4bf*)(sm + OFF_W1 + n * W1_STRIDE + c * 2) = h;
  }
  // W2: [64][512] row-major
  const v4f* src2 = (const v4f*)W2g;
  for (int i = tid; i < (64 * HID) / 4; i += NTHR) {
    int e = i * 4;
    int n = e / HID, c = e % HID;
    v4f f = src2[i];
    v4bf h = { (bf16_t)f[0], (bf16_t)f[1], (bf16_t)f[2], (bf16_t)f[3] };
    *(v4bf*)(sm + OFF_W2 + n * W2_STRIDE + c * 2) = h;
  }
  for (int i = tid; i < HID; i += NTHR) s_b1[i] = b1g[i];
  if (tid < 64) s_b2[tid] = b2g[tid];
}

// One fused MLP (160 -> 512 -> 64) for this wave's 16-row tile, plus the
// coupling epilogue: le = 0.636*atan(r[:, :32]); y = exp(le)*x_mul + r[:, 32:].
__device__ __forceinline__ void run_mlp(char* sm, int m0, int lane,
                                        const float* s_b1, const float* s_b2,
                                        float* s_logj, int xcol, int writeAct) {
  char*  act  = sm + OFF_ACT;
  char*  w1   = sm + OFF_W1;
  char*  w2   = sm + OFF_W2;
  char*  hs   = sm + OFF_HS + (m0 >> 4) * 1024;  // per-wave scratch
  float* xlds = (float*)(sm + OFF_X);
  const int lr = lane & 15;     // row within M-tile (A), column (B/C layouts)
  const int lh = lane >> 4;     // half-wave select

  // --- A fragments: 5 x (16x32 bf16) from [x_half | H] ---
  Frag a[5];
#pragma unroll
  for (int kk = 0; kk < 5; ++kk) {
    const char* p = act + (m0 + lr) * ACT_STRIDE + (kk * 32 + lh * 8) * 2;
    a[kk].q[0] = *(const v4u*)p;          // K = kb .. kb+7
    a[kk].q[1] = *(const v4u*)(p + 32);   // K = kb+16 .. kb+23
  }

  // --- output accumulators (16x64), initialized with b2 ---
  v8f acc[4];
#pragma unroll
  for (int t = 0; t < 4; ++t) {
    float bv = s_b2[t * 16 + lr];
#pragma unroll
    for (int j = 0; j < 8; ++j) acc[t][j] = bv;
  }

  // --- fused GEMM1 (160->512) + ReLU + GEMM2 (512->64) over 32-wide hidden chunks ---
  for (int nh = 0; nh < 16; ++nh) {
    const int hb = nh * 32;
    v8f h0, h1;
    {
      float b0 = s_b1[hb + lr], b1v = s_b1[hb + 16 + lr];
#pragma unroll
      for (int j = 0; j < 8; ++j) { h0[j] = b0; h1[j] = b1v; }
    }
#pragma unroll
    for (int kk = 0; kk < 5; ++kk) {
      Frag bf0, bf1;
      const char* p0 = w1 + (hb + lr) * W1_STRIDE + (kk * 32 + lh * 8) * 2;
      const char* p1 = w1 + (hb + 16 + lr) * W1_STRIDE + (kk * 32 + lh * 8) * 2;
      bf0.q[0] = *(const v4u*)p0; bf0.q[1] = *(const v4u*)(p0 + 32);
      bf1.q[0] = *(const v4u*)p1; bf1.q[1] = *(const v4u*)(p1 + 32);
      h0 = wmma_bf16(a[kk].v, bf0.v, h0);
      h1 = wmma_bf16(a[kk].v, bf1.v, h1);
    }
    // ReLU + C/D->A relayout through per-wave LDS scratch (16x32 bf16, 64B row stride)
#pragma unroll
    for (int j = 0; j < 8; ++j) {
      int r = lh * 8 + j;
      bf16_t* row = (bf16_t*)(hs + r * 64);
      row[lr]      = (bf16_t)fmaxf(h0[j], 0.0f);
      row[16 + lr] = (bf16_t)fmaxf(h1[j], 0.0f);
    }
    Frag a2;
    {
      const char* p = hs + lr * 64 + (lh * 8) * 2;
      a2.q[0] = *(const v4u*)p;
      a2.q[1] = *(const v4u*)(p + 32);
    }
#pragma unroll
    for (int t = 0; t < 4; ++t) {
      Frag b2f;
      const char* p = w2 + (t * 16 + lr) * W2_STRIDE + (hb + lh * 8) * 2;
      b2f.q[0] = *(const v4u*)p; b2f.q[1] = *(const v4u*)(p + 32);
      acc[t] = wmma_bf16(a2.v, b2f.v, acc[t]);
    }
  }

  // --- coupling epilogue + log-Jacobian partials ---
#pragma unroll
  for (int j = 0; j < 8; ++j) {
    int m = m0 + lh * 8 + j;
    float lsum = 0.0f;
#pragma unroll
    for (int t = 0; t < 2; ++t) {
      int   c  = t * 16 + lr;                 // column in [0,32)
      float s  = acc[t][j];
      float le = 0.636f * atanf(s);           // CLAMP = 1
      float e  = __expf(le);
      float xv = xlds[m * D_FEAT + xcol + c];
      float y  = e * xv + acc[t + 2][j];
      xlds[m * D_FEAT + xcol + c] = y;
      if (writeAct)
        *(bf16_t*)(sm + OFF_ACT + m * ACT_STRIDE + c * 2) = (bf16_t)y;
      lsum += le;
    }
    atomicAdd(&s_logj[m], lsum);              // ds_add_f32
  }
}

__global__ __launch_bounds__(NTHR, 1)
void cinn_flow_kernel(const float* __restrict__ qf, const float* __restrict__ Hg,
                      const int* __restrict__ perms,
                      const float* __restrict__ an_ls, const float* __restrict__ an_bi,
                      const float* __restrict__ s1W1, const float* __restrict__ s1b1,
                      const float* __restrict__ s1W2, const float* __restrict__ s1b2,
                      const float* __restrict__ s2W1, const float* __restrict__ s2b1,
                      const float* __restrict__ s2W2, const float* __restrict__ s2b2,
                      float* __restrict__ out, int Btot) {
  extern __shared__ char sm[];
  __shared__ float s_b1[HID];
  __shared__ float s_b2[64];
  __shared__ float s_els[64];
  __shared__ float s_bi[64];
  __shared__ int   s_perm[64];
  __shared__ float s_logj[BT];
  __shared__ float s_sls;

  const int tid     = threadIdx.x;
  const int lane    = tid & 31;
  const int wave    = tid >> 5;
  const int m0      = wave * 16;
  const int rowbase = blockIdx.x * BT;

  float* xlds = (float*)(sm + OFF_X);
  char*  act  = sm + OFF_ACT;

  // ---- init: logj, x tile, bf16 copy of H (constant across blocks) ----
  if (tid < BT) s_logj[tid] = 0.0f;
  if (tid == 0) s_sls = 0.0f;
  for (int i = tid; i < BT * D_FEAT; i += NTHR)
    xlds[i] = qf[rowbase * D_FEAT + i];
  for (int i = tid; i < BT * DC; i += NTHR) {
    int r = i >> 7, c = i & (DC - 1);
    *(bf16_t*)(act + r * ACT_STRIDE + (L1 + c) * 2) = (bf16_t)Hg[(rowbase + r) * DC + c];
  }
  __syncthreads();

  for (int k = 0; k < KBLK; ++k) {
    // ---- stage per-block permutation/affine params + s2 (first coupling) weights ----
    if (tid < 64) {
      float ls   = an_ls[k * 64 + tid];
      s_els[tid] = __expf(ls);
      s_bi[tid]  = an_bi[k * 64 + tid];
      s_perm[tid] = perms[k * 64 + tid];
      atomicAdd(&s_sls, ls);
    }
    stage_weights(sm, s2W1 + (size_t)k * HID * NIN, s2b1 + k * HID,
                  s2W2 + (size_t)k * 64 * HID, s2b2 + k * 64, s_b1, s_b2, tid);
    if (tid == 0) {  // warm L2 path for this block's second-coupling weights
      __builtin_prefetch(s1W1 + (size_t)k * HID * NIN, 0, 1);
      __builtin_prefetch(s1W2 + (size_t)k * 64 * HID, 0, 1);
    }
    __syncthreads();

    // ---- permute + affine normalization (wave-local rows), fill act x2 cols ----
    float tmp[32];
#pragma unroll
    for (int i = 0; i < 32; ++i) {
      int idx = i * 32 + lane;
      int m = m0 + (idx >> 6), d = idx & 63;
      tmp[i] = xlds[m * D_FEAT + s_perm[d]];
    }
#pragma unroll
    for (int i = 0; i < 32; ++i) {
      int idx = i * 32 + lane;
      int m = m0 + (idx >> 6), d = idx & 63;
      float v = tmp[i] * s_els[d] + s_bi[d];
      xlds[m * D_FEAT + d] = v;
      if (d >= L1)  // x2 -> activation input of first MLP
        *(bf16_t*)(act + m * ACT_STRIDE + (d - L1) * 2) = (bf16_t)v;
    }

    // ---- MLP s2 on [x2, H]: y1 = exp(le2)*x1 + add; writes x cols [0,32) + act ----
    run_mlp(sm, m0, lane, s_b1, s_b2, s_logj, /*xcol=*/0, /*writeAct=*/1);
    __syncthreads();

    // ---- stage s1 (second coupling) weights ----
    stage_weights(sm, s1W1 + (size_t)k * HID * NIN, s1b1 + k * HID,
                  s1W2 + (size_t)k * 64 * HID, s1b2 + k * 64, s_b1, s_b2, tid);
    __syncthreads();

    // ---- MLP s1 on [y1, H]: y2 = exp(le1)*x2 + add; writes x cols [32,64) ----
    run_mlp(sm, m0, lane, s_b1, s_b2, s_logj, /*xcol=*/L1, /*writeAct=*/0);
    __syncthreads();
  }

  // ---- outputs: z then logj (tuple order) ----
  for (int i = tid; i < BT * D_FEAT; i += NTHR)
    out[(size_t)rowbase * D_FEAT + i] = xlds[i];
  float* logj_out = out + (size_t)Btot * D_FEAT;
  if (tid < BT) logj_out[rowbase + tid] = s_logj[tid] + s_sls;
}

extern "C" void kernel_launch(void* const* d_in, const int* in_sizes, int n_in,
                              void* d_out, int out_size, void* d_ws, size_t ws_size,
                              hipStream_t stream) {
  const float* qf    = (const float*)d_in[0];
  const float* Hg    = (const float*)d_in[1];
  const int*   perms = (const int*)d_in[2];
  const float* anls  = (const float*)d_in[3];
  const float* anbi  = (const float*)d_in[4];
  const float* s1W1  = (const float*)d_in[5];
  const float* s1b1  = (const float*)d_in[6];
  const float* s1W2  = (const float*)d_in[7];
  const float* s1b2  = (const float*)d_in[8];
  const float* s2W1  = (const float*)d_in[9];
  const float* s2b1  = (const float*)d_in[10];
  const float* s2W2  = (const float*)d_in[11];
  const float* s2b2  = (const float*)d_in[12];
  float* out = (float*)d_out;

  const int Btot = in_sizes[0] / D_FEAT;   // 32768
  const int grid = Btot / BT;              // 256 workgroups

  (void)hipFuncSetAttribute((const void*)cinn_flow_kernel,
                            hipFuncAttributeMaxDynamicSharedMemorySize, SMEM_DYN);
  cinn_flow_kernel<<<grid, NTHR, SMEM_DYN, stream>>>(
      qf, Hg, perms, anls, anbi,
      s1W1, s1b1, s1W2, s1b2,
      s2W1, s2b1, s2W2, s2b2,
      out, Btot);
}
